// SelectiveStateSpace_60533269070478
// MI455X (gfx1250) — compile-verified, matrix operand order checked by reading, not run on hardware
//
#include <hip/hip_runtime.h>
#include <hip/hip_bf16.h>
#include <math.h>

// ---------------------------------------------------------------------------
// Problem constants (match reference): B=2, L=1024, DM=1024, DI=2048, N=16,
// K_conv=4, R=64.
// ---------------------------------------------------------------------------
#define BB   2
#define LL   1024
#define DM   1024
#define DI   2048
#define NS   16          // d_state
#define KC   4           // d_conv
#define RR   64          // dt_rank
#define ML   (BB * LL)   // 2048 rows for all GEMMs

typedef __attribute__((ext_vector_type(16))) __bf16 v16bf;
typedef __attribute__((ext_vector_type(8)))  float  v8f;

// ---------------------------------------------------------------------------
// Generic bf16 WMMA GEMM:  C(MxN,f32) = A(MxK,f32) * Bw(KxN,f32)
// Each wave owns a 64x16 output strip (4 stacked 16x16 tiles): the strided
// (column-major) B fragment is loaded ONCE per k-step and reused by 4 WMMAs
// against 4 per-lane-contiguous A fragments. 8 waves / 256 threads per block.
// K must be a multiple of 32; strip count must be a multiple of 8.
// EPI==0: plain store.  EPI==1: C = softplus(C + bias[n])  (dt projection).
// Fragment layouts per CDNA5 ISA 7.12.2:
//   A 16x32 bf16 : lane<16 -> M=lane, elems 0..7=K0..7,  8..15=K16..23
//                  lane>=16-> M=lane-16, elems 0..7=K8..15, 8..15=K24..31
//   B 32x16 bf16 : mirrored (lane selects N column)
//   C/D 16x16 f32: VGPR v -> M = v + 8*(lane>=16), N = lane&15
// ---------------------------------------------------------------------------
template <int EPI>
__global__ void wmma_gemm_bf16(const float* __restrict__ A, int lda,
                               const float* __restrict__ Bw, int ldb,
                               float* __restrict__ C, int ldc,
                               int numTilesN, int K,
                               const float* __restrict__ bias) {
  const int wave  = threadIdx.x >> 5;
  const int lane  = threadIdx.x & 31;
  const int strip = blockIdx.x * 8 + wave;   // 64-row x 16-col strip
  const int sm    = strip / numTilesN;
  const int tn    = strip - sm * numTilesN;
  const int half  = lane >> 4;               // 0: lanes 0-15, 1: lanes 16-31
  const int r     = lane & 15;
  const int m0    = sm * 64;
  const int ncol  = tn * 16 + r;

  const float* __restrict__ Arow0 = A + (size_t)(m0 + r) * lda;

  v8f cacc[4] = {};
  for (int k0 = 0; k0 < K; k0 += 32) {
    const int kb = k0 + half * 8;

    v16bf b;                                 // strided B fragment: load once
#pragma unroll
    for (int i = 0; i < 8; ++i) {
      b[i]     = (__bf16)Bw[(size_t)(kb + i) * ldb + ncol];
      b[i + 8] = (__bf16)Bw[(size_t)(kb + 16 + i) * ldb + ncol];
    }

#pragma unroll
    for (int mt = 0; mt < 4; ++mt) {         // reuse B across 4 M-tiles
      const float* __restrict__ Ar = Arow0 + (size_t)(mt * 16) * lda;
      v16bf a;
#pragma unroll
      for (int i = 0; i < 8; ++i) {
        a[i]     = (__bf16)Ar[kb + i];
        a[i + 8] = (__bf16)Ar[kb + 16 + i];
      }
      // (neg_a, A, neg_b, B, c_mod, C, reuse_a, reuse_b)
      cacc[mt] = __builtin_amdgcn_wmma_f32_16x16x32_bf16(
          false, a, false, b, (short)0, cacc[mt], false, false);
    }
  }

#pragma unroll
  for (int mt = 0; mt < 4; ++mt) {
    const int mbase = m0 + mt * 16 + half * 8;
#pragma unroll
    for (int v = 0; v < 8; ++v) {
      float val = cacc[mt][v];
      if constexpr (EPI == 1) {
        val += bias[ncol];
        val = (val > 20.0f) ? val : log1pf(__expf(val));  // softplus
      }
      C[(size_t)(mbase + v) * ldc + ncol] = val;
    }
  }
}

// ---------------------------------------------------------------------------
// Depthwise causal conv1d (K=4) + SiLU over xs = xz[:, :DI].
// Also extracts conv_cache[b,d,j] = xs[b, L-KC+j, d].
// One thread per (b,l,d) element.
// ---------------------------------------------------------------------------
__global__ void conv_silu_kernel(const float* __restrict__ xz,
                                 const float* __restrict__ conv_w,
                                 const float* __restrict__ conv_b,
                                 float* __restrict__ x_conv,
                                 float* __restrict__ conv_cache) {
  const int idx = blockIdx.x * blockDim.x + threadIdx.x;  // B*L*DI total
  const int d  = idx & (DI - 1);
  const int bl = idx / DI;
  const int l  = bl & (LL - 1);

  float acc = conv_b[d];
#pragma unroll
  for (int j = 0; j < KC; ++j) {
    const int ls = l - (KC - 1) + j;
    const float xv = (ls >= 0) ? xz[((size_t)(bl + ls - l)) * (2 * DI) + d] : 0.0f;
    acc = fmaf(conv_w[d * KC + j], xv, acc);
  }
  acc = acc / (1.0f + __expf(-acc));  // SiLU
  x_conv[(size_t)bl * DI + d] = acc;

  if (l >= LL - KC) {
    const int b = bl / LL;
    conv_cache[((size_t)b * DI + d) * KC + (l - (LL - KC))] =
        xz[(size_t)bl * (2 * DI) + d];
  }
}

// ---------------------------------------------------------------------------
// Selective scan: one thread per (b, d) channel; 16-state in registers,
// sequential over L. Fused epilogue: yz = (y + D*u) * silu(z).
// B_t / C_t (32 floats per (b,t)) broadcast from L2 (192 MB; all operands fit).
// ---------------------------------------------------------------------------
__global__ void scan_kernel(const float* __restrict__ dt,
                            const float* __restrict__ x_dbl,
                            const float* __restrict__ x_conv,
                            const float* __restrict__ xz,
                            const float* __restrict__ A_log,
                            const float* __restrict__ D_param,
                            float* __restrict__ yz,
                            float* __restrict__ h_out) {
  const int d = blockIdx.x * blockDim.x + threadIdx.x;  // 0..DI-1
  const int b = blockIdx.y;                             // 0..B-1

  float Aneg[NS];
#pragma unroll
  for (int n = 0; n < NS; ++n) Aneg[n] = -__expf(A_log[(size_t)d * NS + n]);
  const float Dv = D_param[d];

  float h[NS];
#pragma unroll
  for (int n = 0; n < NS; ++n) h[n] = 0.0f;

  for (int t = 0; t < LL; ++t) {
    const size_t bl = (size_t)b * LL + t;
    const float dtv = dt[bl * DI + d];
    const float u   = x_conv[bl * DI + d];
    const float* __restrict__ xd = x_dbl + bl * (RR + 2 * NS);

    if (t + 1 < LL) {  // global_prefetch_b8 for next timestep's channel data
      __builtin_prefetch(&dt[(bl + 1) * DI + d], 0, 1);
      __builtin_prefetch(&x_conv[(bl + 1) * DI + d], 0, 1);
    }

    float y = 0.0f;
#pragma unroll
    for (int n = 0; n < NS; ++n) {
      const float Bn = xd[RR + n];
      const float Cn = xd[RR + NS + n];
      const float ab = __expf(dtv * Aneg[n]);
      h[n] = fmaf(ab, h[n], dtv * Bn * u);
      y = fmaf(h[n], Cn, y);
    }
    const float z  = xz[bl * (2 * DI) + DI + d];
    const float sz = z / (1.0f + __expf(-z));      // silu(z)
    yz[bl * DI + d] = (y + Dv * u) * sz;
  }

#pragma unroll
  for (int n = 0; n < NS; ++n)
    h_out[((size_t)b * DI + d) * NS + n] = h[n];
}

// ---------------------------------------------------------------------------
// Launcher. Inputs (setup_inputs order):
//  0:x 1:W_in 2:conv_w 3:conv_b 4:W_x 5:W_dt 6:b_dt 7:A_log 8:D_param 9:W_out
// Outputs (flat, return order): out (B,L,DM) | h_final (B,DI,NS) | conv_cache (B,DI,KC)
// Workspace (floats): xz 8.39M | x_conv 4.19M | x_dbl 0.20M | dt 4.19M | yz 4.19M
//  => ~81 MB total.
// ---------------------------------------------------------------------------
extern "C" void kernel_launch(void* const* d_in, const int* /*in_sizes*/, int /*n_in*/,
                              void* d_out, int /*out_size*/, void* d_ws, size_t /*ws_size*/,
                              hipStream_t stream) {
  const float* x       = (const float*)d_in[0];
  const float* W_in    = (const float*)d_in[1];
  const float* conv_w  = (const float*)d_in[2];
  const float* conv_b  = (const float*)d_in[3];
  const float* W_x     = (const float*)d_in[4];
  const float* W_dt    = (const float*)d_in[5];
  const float* b_dt    = (const float*)d_in[6];
  const float* A_log   = (const float*)d_in[7];
  const float* D_param = (const float*)d_in[8];
  const float* W_out   = (const float*)d_in[9];

  float* out        = (float*)d_out;                       // (B,L,DM)
  float* h_final    = out + (size_t)ML * DM;               // (B,DI,NS)
  float* conv_cache = h_final + (size_t)BB * DI * NS;      // (B,DI,KC)

  float* ws     = (float*)d_ws;
  float* xz     = ws;                                      // (ML, 2*DI)
  float* x_conv = xz + (size_t)ML * 2 * DI;                // (ML, DI)
  float* x_dbl  = x_conv + (size_t)ML * DI;                // (ML, RR+2*NS)
  float* dtbuf  = x_dbl + (size_t)ML * (RR + 2 * NS);      // (ML, DI)
  float* yz     = dtbuf + (size_t)ML * DI;                 // (ML, DI)

  // Strips per GEMM: (M/64) * (N/16); 8 waves per block.
  // 1) xz = x @ W_in          M=2048 K=1024 N=4096  -> 32*256 strips
  wmma_gemm_bf16<0><<<(ML / 64) * (2 * DI / 16) / 8, 256, 0, stream>>>(
      x, DM, W_in, 2 * DI, xz, 2 * DI, (2 * DI) / 16, DM, nullptr);

  // 2) causal depthwise conv + SiLU; extract conv_cache
  conv_silu_kernel<<<(BB * LL * DI) / 256, 256, 0, stream>>>(
      xz, conv_w, conv_b, x_conv, conv_cache);

  // 3) x_dbl = x_conv @ W_x   M=2048 K=2048 N=96    -> 32*6 strips
  wmma_gemm_bf16<0><<<(ML / 64) * ((RR + 2 * NS) / 16) / 8, 256, 0, stream>>>(
      x_conv, DI, W_x, RR + 2 * NS, x_dbl, RR + 2 * NS, (RR + 2 * NS) / 16, DI,
      nullptr);

  // 4) dt = softplus(x_dbl[:, :R] @ W_dt + b_dt)   M=2048 K=64 N=2048
  wmma_gemm_bf16<1><<<(ML / 64) * (DI / 16) / 8, 256, 0, stream>>>(
      x_dbl, RR + 2 * NS, W_dt, DI, dtbuf, DI, DI / 16, RR, b_dt);

  // 5) selective scan + fused (+D*u) * silu(z)
  {
    dim3 grid(DI / 256, BB);
    scan_kernel<<<grid, 256, 0, stream>>>(dtbuf, x_dbl, x_conv, xz, A_log,
                                          D_param, yz, h_final);
  }

  // 6) out = yz @ W_out       M=2048 K=2048 N=1024  -> 32*64 strips
  wmma_gemm_bf16<0><<<(ML / 64) * (DM / 16) / 8, 256, 0, stream>>>(
      yz, DI, W_out, DM, out, DM, DM / 16, DI, nullptr);
}
